// Model_28879360098823
// MI455X (gfx1250) — matrix operand order repaired and checked by reading, not compiled, not run on hardware
//
#include <hip/hip_runtime.h>

// Problem dims
#define SEQ   512
#define BATCH 64
#define EMB   256
#define HID   512
#define OUTD  2
#define NWG   8      // persistent workgroups in recurrence; each owns 64 hidden cols

typedef __attribute__((ext_vector_type(16))) __bf16 v16bf;
typedef __attribute__((ext_vector_type(8)))  float  v8f;
typedef __attribute__((ext_vector_type(4)))  unsigned int u32x4;
typedef __attribute__((ext_vector_type(8)))  int          i32x8;
typedef __attribute__((ext_vector_type(4)))  int          i32x4;

union Frag {
  v16bf v;
  uint4 q[2];
};

static __device__ __forceinline__ v8f wmma_bf16(v16bf a, v16bf b, v8f c) {
  // (neg_a, A, neg_b, B, c_mod, C, reuse_a, reuse_b)
  return __builtin_amdgcn_wmma_f32_16x16x32_bf16(false, a, false, b, (short)0, c,
                                                 false, false);
}

// Device-wide split barrier for the persistent recurrence kernel.
static __device__ __forceinline__ void grid_sync(unsigned int* cnt,
                                                 unsigned int target, int tid) {
  __threadfence();          // release this block's h writes to device scope
  __syncthreads();
  if (tid == 0) {
    __hip_atomic_fetch_add(cnt, 1u, __ATOMIC_RELEASE, __HIP_MEMORY_SCOPE_AGENT);
    while (__hip_atomic_load(cnt, __ATOMIC_ACQUIRE, __HIP_MEMORY_SCOPE_AGENT) < target) {
      __builtin_amdgcn_s_sleep(2);
    }
  }
  __syncthreads();
  __threadfence();          // acquire other blocks' writes
}

// ---------------------------------------------------------------------------
// Kernel A: xp[m, n] = emb[text[m]] @ W_ih^T + b_ih   (m = t*BATCH + b)
// M = 32768, N = 512, K = 256.  Block: 256 threads (8 waves), tile 64x64.
// ---------------------------------------------------------------------------
__global__ __launch_bounds__(256) void xproj_kernel(
    const int* __restrict__ text, const float* __restrict__ emb,
    const float* __restrict__ W_ih, const float* __restrict__ b_ih,
    float* __restrict__ xp)
{
  __shared__ __align__(16) __bf16 lds_a[64 * EMB];  // [row][k]  32 KB
  __shared__ __align__(16) __bf16 lds_b[64 * EMB];  // [n_local][k] 32 KB

  const int tid = threadIdx.x;
  const int n0  = blockIdx.x * 64;
  const int m0  = blockIdx.y * 64;

  // Stage A (embedding gather -> bf16) and B (W_ih slice -> bf16). k == tid.
  for (int i = 0; i < 64; ++i) {
    const int tok = text[m0 + i];                       // scalar per row
    lds_a[i * EMB + tid] = (__bf16)emb[tok * EMB + tid];
    lds_b[i * EMB + tid] = (__bf16)W_ih[(n0 + i) * EMB + tid];
  }
  __syncthreads();

  const int wave = tid >> 5, lane = tid & 31;
  const int l16 = lane & 15, half = lane >> 4;
  const int mt  = wave >> 1;           // 0..3  (16-row M tile)
  const int nf0 = (wave & 1) * 2;      // this wave does n-frags nf0, nf0+1

  v8f acc0 = {}, acc1 = {};
  const int arow = mt * 16 + l16;
  const int br0  = (nf0 + 0) * 16 + l16;
  const int br1  = (nf0 + 1) * 16 + l16;

  #pragma unroll
  for (int kc = 0; kc < EMB / 32; ++kc) {
    Frag a, b0, b1;
    // A layout (16x32): lane m=l16; K = base + half*16 + {0..15} contiguous
    const uint4* pa = (const uint4*)&lds_a[arow * EMB + kc * 32 + half * 16];
    a.q[0] = pa[0];
    a.q[1] = pa[1];
    // B layout (32x16): lane n=l16; K = base + half*16 + {0..15} contiguous
    const uint4* pb0 = (const uint4*)&lds_b[br0 * EMB + kc * 32 + half * 16];
    b0.q[0] = pb0[0]; b0.q[1] = pb0[1];
    const uint4* pb1 = (const uint4*)&lds_b[br1 * EMB + kc * 32 + half * 16];
    b1.q[0] = pb1[0]; b1.q[1] = pb1[1];

    acc0 = wmma_bf16(a.v, b0.v, acc0);
    acc1 = wmma_bf16(a.v, b1.v, acc1);
  }

  // Epilogue: +b_ih, store fp32. C/D layout: vgpr r -> m = r + 8*half, n = l16.
  const int n_0 = n0 + (nf0 + 0) * 16 + l16;
  const int n_1 = n0 + (nf0 + 1) * 16 + l16;
  const float bi0 = b_ih[n_0], bi1 = b_ih[n_1];
  #pragma unroll
  for (int r = 0; r < 8; ++r) {
    const int m = m0 + mt * 16 + r + 8 * half;
    xp[(size_t)m * HID + n_0] = acc0[r] + bi0;
    xp[(size_t)m * HID + n_1] = acc1[r] + bi1;
  }
}

// ---------------------------------------------------------------------------
// Kernel B: persistent recurrence.  8 WGs; WG wg owns hidden cols [wg*64, +64).
// W_hh slice lives in LDS (bf16) for all 512 steps.  h ping-pongs through L2
// as bf16; each step one wave per WG pulls the whole 64KB h_t into LDS with a
// single Tensor-Data-Mover op (tensor_load_to_lds), waited on TENSORcnt.
// ---------------------------------------------------------------------------
__global__ __launch_bounds__(256) void rnn_kernel(
    const float* __restrict__ W_hh, const float* __restrict__ b_hh,
    const float* __restrict__ xp, __bf16* __restrict__ hbuf0,
    __bf16* __restrict__ hbuf1, unsigned int* __restrict__ sync_cnt)
{
  __shared__ __align__(16) __bf16 lds_bhh[64 * HID]; // [n_local][k] 64 KB, resident
  __shared__ __align__(16) __bf16 lds_h[BATCH * HID];// [b][k]       64 KB, per step

  const int tid = threadIdx.x;
  const int wg  = blockIdx.x;
  const int n0  = wg * 64;

  // Preload this WG's W_hh slice as bf16 (B(k,n) = W_hh[n][k]).
  for (int i = 0; i < (64 * HID) / 256; ++i) {
    const int idx = i * 256 + tid;
    const int n = idx >> 9, k = idx & (HID - 1);
    lds_bhh[n * HID + k] = (__bf16)W_hh[(size_t)(n0 + n) * HID + k];
  }
  // h0 = 0 in buffer 0 (our slice), bf16.
  for (int i = 0; i < (BATCH * 64) / 256; ++i) {
    const int idx = i * 256 + tid;
    const int b = idx >> 6, n = idx & 63;
    hbuf0[b * HID + n0 + n] = (__bf16)0.0f;
  }
  __syncthreads();
  grid_sync(sync_cnt, 1u * NWG, tid);

  // ---- TDM descriptors (ISA cdna5 §8): 64KB copy = one row of 8192 x 8B ----
  const unsigned ldsoff = (unsigned)(unsigned long long)&lds_h[0];
  u32x4 g0_even, g0_odd;
  {
    unsigned long long ga0 = (unsigned long long)hbuf0;
    unsigned long long ga1 = (unsigned long long)hbuf1;
    g0_even[0] = 1u;                                   // count = 1 (valid)
    g0_even[1] = ldsoff;                               // lds_addr
    g0_even[2] = (unsigned)(ga0 & 0xffffffffull);      // global_addr[31:0]
    g0_even[3] = (unsigned)((ga0 >> 32) & 0x01ffffffull) | (2u << 30); // type=2
    g0_odd[0]  = 1u;
    g0_odd[1]  = ldsoff;
    g0_odd[2]  = (unsigned)(ga1 & 0xffffffffull);
    g0_odd[3]  = (unsigned)((ga1 >> 32) & 0x01ffffffull) | (2u << 30);
  }
  i32x8 g1;
  g1[0] = (int)(3u << 16);          // workgroup_mask=0, data_size=3 (8 bytes)
  g1[1] = (int)(0x2000u << 16);     // tensor_dim0[15:0] = 8192 (bits 79:48)
  g1[2] = (int)(1u << 16);          // tensor_dim0[31:16]=0, tensor_dim1[15:0]=1
  g1[3] = (int)(0x2000u << 16);     // tensor_dim1 hi=0, tile_dim0 = 8192
  g1[4] = 1;                        // tile_dim1 = 1, tile_dim2 = 0
  g1[5] = 0x2000;                   // tensor_dim0_stride = 8192
  g1[6] = (int)(0x2000u << 16);     // dim0_stride hi=0, tensor_dim1_stride lo
  g1[7] = 0;
  i32x4 gz4 = {0, 0, 0, 0};         // groups 2/3 unused (<=2D tensor)
  i32x8 gz8 = {0, 0, 0, 0, 0, 0, 0, 0};

  const int wave = tid >> 5, lane = tid & 31;
  const int l16 = lane & 15, half = lane >> 4;
  const int mt  = wave >> 1;            // batch tile 0..3
  const int nf0 = (wave & 1) * 2;

  const int n_0 = n0 + (nf0 + 0) * 16 + l16;
  const int n_1 = n0 + (nf0 + 1) * 16 + l16;
  const float bh0 = b_hh[n_0], bh1 = b_hh[n_1];   // hoisted across all steps

  __bf16* bufs[2] = {hbuf0, hbuf1};

  for (int t = 0; t < SEQ; ++t) {
    __bf16* hdst = bufs[(t + 1) & 1];
    const float* xpt = xp + (size_t)t * BATCH * HID;

    // One TDM op pulls all of h_t (bf16, 64KB) from L2 into LDS.
    if (tid < 32) {
      u32x4 g0 = (t & 1) ? g0_odd : g0_even;
      __builtin_amdgcn_tensor_load_to_lds(g0, g1, gz4, gz4, gz8, 0);
      __builtin_amdgcn_s_wait_tensorcnt(0);
    }
    // Prefetch next step's xp tile (global_prefetch_b8).
    if (t + 1 < SEQ)
      __builtin_prefetch((const char*)xpt + (size_t)BATCH * HID * 4 + tid * 512, 0, 0);
    __syncthreads();

    // acc starts at xp (already contains b_ih).
    v8f acc0, acc1;
    #pragma unroll
    for (int r = 0; r < 8; ++r) {
      const int b = mt * 16 + r + 8 * half;
      acc0[r] = xpt[(size_t)b * HID + n_0];
      acc1[r] = xpt[(size_t)b * HID + n_1];
    }

    const int arow = mt * 16 + l16;
    const int br0  = (nf0 + 0) * 16 + l16;
    const int br1  = (nf0 + 1) * 16 + l16;
    #pragma unroll
    for (int kc = 0; kc < HID / 32; ++kc) {
      Frag a, b0, b1;
      const uint4* pa = (const uint4*)&lds_h[arow * HID + kc * 32 + half * 16];
      a.q[0] = pa[0];
      a.q[1] = pa[1];
      const uint4* pb0 = (const uint4*)&lds_bhh[br0 * HID + kc * 32 + half * 16];
      b0.q[0] = pb0[0]; b0.q[1] = pb0[1];
      const uint4* pb1 = (const uint4*)&lds_bhh[br1 * HID + kc * 32 + half * 16];
      b1.q[0] = pb1[0]; b1.q[1] = pb1[1];

      acc0 = wmma_bf16(a.v, b0.v, acc0);
      acc1 = wmma_bf16(a.v, b1.v, acc1);
    }

    // h_{t+1} = tanh(acc + b_hh) -> other buffer, stored bf16 (WMMA A input
    // precision anyway), stays hot in L2.
    #pragma unroll
    for (int r = 0; r < 8; ++r) {
      const int b = mt * 16 + r + 8 * half;
      hdst[(size_t)b * HID + n_0] = (__bf16)tanhf(acc0[r] + bh0);
      hdst[(size_t)b * HID + n_1] = (__bf16)tanhf(acc1[r] + bh1);
    }
    __syncthreads();                       // lds_h reuse + writes done
    grid_sync(sync_cnt, (unsigned)(t + 2) * NWG, tid);
  }
}

// ---------------------------------------------------------------------------
// Kernel C: out[b,o] = h_final[b,:] . W_fc[o,:] + b_fc[o]   (64x2 — tiny)
// ---------------------------------------------------------------------------
__global__ __launch_bounds__(128) void fc_kernel(
    const __bf16* __restrict__ h, const float* __restrict__ W_fc,
    const float* __restrict__ b_fc, float* __restrict__ out)
{
  const int tid = threadIdx.x;           // 0..127
  const int b = tid >> 1, o = tid & 1;
  float acc = b_fc[o];
  for (int k = 0; k < HID; ++k)
    acc += (float)h[b * HID + k] * W_fc[o * HID + k];
  out[b * OUTD + o] = acc;
}

extern "C" void kernel_launch(void* const* d_in, const int* in_sizes, int n_in,
                              void* d_out, int out_size, void* d_ws, size_t ws_size,
                              hipStream_t stream) {
  const int*   text = (const int*)  d_in[0];
  const float* emb  = (const float*)d_in[1];
  const float* W_ih = (const float*)d_in[2];
  const float* W_hh = (const float*)d_in[3];
  const float* b_ih = (const float*)d_in[4];
  const float* b_hh = (const float*)d_in[5];
  const float* W_fc = (const float*)d_in[6];
  const float* b_fc = (const float*)d_in[7];
  float* out = (float*)d_out;

  char* ws = (char*)d_ws;
  unsigned int* cnt = (unsigned int*)ws;                         // 256 B (reset below)
  __bf16* hbuf0 = (__bf16*)(ws + 256);                           // 64 KB
  __bf16* hbuf1 = (__bf16*)(ws + 256 + BATCH * HID * 2);         // 64 KB
  float*  xp    = (float*) (ws + 256 + 2 * BATCH * HID * 2);     // 64 MB

  (void)hipMemsetAsync(cnt, 0, 256, stream);                     // graph-capture safe

  xproj_kernel<<<dim3(HID / 64, (SEQ * BATCH) / 64), 256, 0, stream>>>(
      text, emb, W_ih, b_ih, xp);

  rnn_kernel<<<dim3(NWG), 256, 0, stream>>>(W_hh, b_hh, xp, hbuf0, hbuf1, cnt);

  // t = 511 wrote bufs[(512)&1] == hbuf0.
  fc_kernel<<<dim3(1), 128, 0, stream>>>(hbuf0, W_fc, b_fc, out);
}
